// Model_23596550324504
// MI455X (gfx1250) — matrix operand order repaired and checked by reading, not compiled, not run on hardware
//
#include <hip/hip_runtime.h>

// ---------------------------------------------------------------------------
// MI455X / gfx1250, wave32.  WMMA: v_wmma_f32_16x16x32_bf16.
// Model: conv(3x3)+pool -> conv(3x3)+pool -> two MLP branches (B=16).
// Big FCs are weight-streaming bound (1.76 GB fp32 @ 23.3 TB/s): stream fp32
// once, convert to bf16 in registers (native cvt), M=16 (batch) per WMMA tile.
// ---------------------------------------------------------------------------

typedef __attribute__((ext_vector_type(16))) __bf16 v16bf;
typedef __attribute__((ext_vector_type(8)))  __bf16 v8bf;
typedef __attribute__((ext_vector_type(2)))  __bf16 v2bf;
typedef __attribute__((ext_vector_type(8)))  float  v8f;

#define BATCH 16
#define FLATN 571520          // 47*190*64

#if defined(__has_builtin)
#  if __has_builtin(__builtin_amdgcn_cvt_pk_bf16_f32)
#    define HAVE_PK_BF16 1
#  endif
#endif

__device__ __forceinline__ int imin(int a, int b) { return a < b ? a : b; }
__device__ __forceinline__ float leakyf(float x) { return x >= 0.f ? x : 0.01f * x; }

// Packed f32x2 -> bf16x2 (native v_cvt_pk_bf16_f32 when available, RNE)
__device__ __forceinline__ v2bf cvt2(float lo, float hi) {
#ifdef HAVE_PK_BF16
    auto r = __builtin_amdgcn_cvt_pk_bf16_f32(lo, hi);
    return __builtin_bit_cast(v2bf, r);
#else
    v2bf r; r[0] = (__bf16)lo; r[1] = (__bf16)hi; return r;
#endif
}

union V16 { v16bf v; v2bf p[8]; __bf16 e[16]; };
union V8x2 { v16bf v; struct { v8bf lo, hi; } s; };

__device__ __forceinline__ void zero16(V16& a) {
#pragma unroll
    for (int i = 0; i < 8; ++i) a.p[i] = cvt2(0.f, 0.f);
}

__device__ __forceinline__ v8f zero8f() {
    v8f z;
#pragma unroll
    for (int i = 0; i < 8; ++i) z[i] = 0.f;
    return z;
}

__device__ __forceinline__ v8f wmma_bf16(v16bf a, v16bf b, v8f c) {
    return __builtin_amdgcn_wmma_f32_16x16x32_bf16(false, a, false, b,
                                                   (short)0, c, false, false);
}

// Two 16B bf16 chunks -> v16bf (callers use contiguous k = 16h + s, B built to match)
__device__ __forceinline__ v16bf load_pair(const __bf16* p0, const __bf16* p1) {
    V8x2 u;
    u.s.lo = *(const v8bf*)p0;
    u.s.hi = *(const v8bf*)p1;
    return u.v;
}

// ---------------------------------------------------------------------------
// conv1 (3x3, 1->32, VALID) + relu + maxpool2, input hidden [16,197,768],
// output stage1 bf16 [16,97,383,32].  One wave = (b, pooled-row, 16-col strip).
// K = 9 taps padded to 32.
// ---------------------------------------------------------------------------
__global__ __launch_bounds__(128) void conv1_pool_kernel(
    const float* __restrict__ hidden, const float* __restrict__ w1,
    const float* __restrict__ b1, __bf16* __restrict__ stage1)
{
    const int XT = 48, PY = 97, TASKS = BATCH * PY * XT;
    int wave = (blockIdx.x * blockDim.x + threadIdx.x) >> 5;
    int lane = threadIdx.x & 31;
    if (wave >= TASKS) return;
    int xt = wave % XT; int t = wave / XT; int py = t % PY; int b = t / PY;
    int h = lane >> 4, ln = lane & 15;

    // B tiles: [9,32] fp32 weights -> 32x16 bf16, K 0..8 valid, rest zero.
    V16 bt[2]; zero16(bt[0]); zero16(bt[1]);
    if (h == 0) {
#pragma unroll
        for (int j = 0; j < 9; ++j) {
            bt[0].e[j] = (__bf16)w1[j * 32 + ln];
            bt[1].e[j] = (__bf16)w1[j * 32 + 16 + ln];
        }
    }

    v8f acc[2][2];
#pragma unroll
    for (int r = 0; r < 2; ++r)
#pragma unroll
        for (int nt = 0; nt < 2; ++nt) acc[r][nt] = zero8f();

    int col = 16 * xt + ln;                    // conv output column (M index)
#pragma unroll
    for (int r2 = 0; r2 < 2; ++r2) {
        int row = 2 * py + r2;                 // conv output row
        V16 a; zero16(a);
        if (h == 0) {                          // K = 0..8
#pragma unroll
            for (int j = 0; j < 9; ++j) {
                int ky = j / 3, kx = j - ky * 3;
                int c = imin(col + kx, 767);
                a.e[j] = (__bf16)hidden[(size_t)(b * 197 + row + ky) * 768 + c];
            }
        } else {                               // tap 8 again; B half 1 is zero, adds 0
            int c = imin(col + 2, 767);
            a.e[0] = (__bf16)hidden[(size_t)(b * 197 + row + 2) * 768 + c];
        }
        acc[r2][0] = wmma_bf16(a.v, bt[0].v, acc[r2][0]);
        acc[r2][1] = wmma_bf16(a.v, bt[1].v, acc[r2][1]);
    }

    // bias + relu + 2x2 maxpool (lane holds M=r and M=r+8 via half h)
#pragma unroll
    for (int nt = 0; nt < 2; ++nt) {
        float bias = b1[nt * 16 + ln];
#pragma unroll
        for (int j = 0; j < 4; ++j) {
            float m0 = fmaxf(acc[0][nt][2 * j], acc[0][nt][2 * j + 1]);
            float m1 = fmaxf(acc[1][nt][2 * j], acc[1][nt][2 * j + 1]);
            float v = fmaxf(fmaxf(m0, m1) + bias, 0.f);
            int px = 8 * xt + h * 4 + j;
            if (px < 383)
                stage1[((size_t)(b * 97 + py) * 383 + px) * 32 + nt * 16 + ln]
                    = (__bf16)v;
        }
    }
}

// ---------------------------------------------------------------------------
// conv2 (3x3, 32->64) + relu + maxpool2: stage1 bf16 -> x bf16 [16,FLATN].
// K = 288 = 9 ksteps of 32 channels (HWIO weights are already [288,64]).
// Per wave: 2 rows x 4 N-tiles x 9 ksteps = 72 WMMAs -> 8x64 pooled outputs.
// ---------------------------------------------------------------------------
__global__ __launch_bounds__(128) void conv2_pool_kernel(
    const __bf16* __restrict__ stage1, const float* __restrict__ w2,
    const float* __restrict__ b2, __bf16* __restrict__ xout)
{
    const int XT = 24, PY = 47, TASKS = BATCH * PY * XT;
    int wave = (blockIdx.x * blockDim.x + threadIdx.x) >> 5;
    int lane = threadIdx.x & 31;
    if (wave >= TASKS) return;
    int xt = wave % XT; int t = wave / XT; int py = t % PY; int b = t / PY;
    int h = lane >> 4, ln = lane & 15;

    v8f acc[2][4];
#pragma unroll
    for (int r = 0; r < 2; ++r)
#pragma unroll
        for (int nt = 0; nt < 4; ++nt) acc[r][nt] = zero8f();

    for (int kq = 0; kq < 9; ++kq) {
        int ky = kq / 3, kx = kq - ky * 3;
        const float* wrow = w2 + (size_t)(kq * 32 + h * 16) * 64 + ln;
        V16 bt[4];
#pragma unroll
        for (int nt = 0; nt < 4; ++nt)
#pragma unroll
            for (int q = 0; q < 8; ++q)
                bt[nt].p[q] = cvt2(wrow[(size_t)(2 * q) * 64 + nt * 16],
                                   wrow[(size_t)(2 * q + 1) * 64 + nt * 16]);

        int c = imin(16 * xt + ln + kx, 382);
#pragma unroll
        for (int r2 = 0; r2 < 2; ++r2) {
            int row = 2 * py + r2 + ky;
            const __bf16* p = stage1 + ((size_t)(b * 97 + row) * 383 + c) * 32;
            v16bf a = load_pair(p + h * 16, p + h * 16 + 8);
#pragma unroll
            for (int nt = 0; nt < 4; ++nt)
                acc[r2][nt] = wmma_bf16(a, bt[nt].v, acc[r2][nt]);
        }
    }

#pragma unroll
    for (int nt = 0; nt < 4; ++nt) {
        float bias = b2[nt * 16 + ln];
#pragma unroll
        for (int j = 0; j < 4; ++j) {
            float m0 = fmaxf(acc[0][nt][2 * j], acc[0][nt][2 * j + 1]);
            float m1 = fmaxf(acc[1][nt][2 * j], acc[1][nt][2 * j + 1]);
            float v = fmaxf(fmaxf(m0, m1) + bias, 0.f);
            int px = 8 * xt + h * 4 + j;
            if (px < 190)
                xout[(size_t)b * FLATN + (size_t)(py * 190 + px) * 64
                     + nt * 16 + ln] = (__bf16)v;
        }
    }
}

// ---------------------------------------------------------------------------
// Seed the big-FC accumulators with their biases.
// ---------------------------------------------------------------------------
__global__ void init_fc1_kernel(const float* __restrict__ b512,
                                const float* __restrict__ b256b,
                                float* __restrict__ y1, float* __restrict__ z1)
{
    int i = blockIdx.x * blockDim.x + threadIdx.x;
    if (i < 16 * 512) y1[i] = b512[i & 511];
    else if (i < 16 * 512 + 16 * 256) {
        int q = i - 16 * 512;
        z1[q] = b256b[q & 255];
    }
}

// ---------------------------------------------------------------------------
// Big FCs: x[16,FLATN] @ w512 and @ w256b, K split across waves, f32 HW
// atomics into bias-seeded buffers.  Weight stream is the HBM bottleneck.
// ---------------------------------------------------------------------------
__global__ __launch_bounds__(256) void fc_big_kernel(
    const __bf16* __restrict__ xact,
    const float* __restrict__ w512, const float* __restrict__ w256b,
    float* __restrict__ y1, float* __restrict__ z1)
{
    const int KSTEPS = FLATN / 32;            // 17860
    const int CH = 128;                       // ksteps per wave
    const int NCH = (KSTEPS + CH - 1) / CH;   // 140
    int wave = (blockIdx.x * blockDim.x + threadIdx.x) >> 5;
    int lane = threadIdx.x & 31;
    int t = wave / NCH, c = wave - t * NCH;
    if (t >= 48) return;
    int ks0 = c * CH, ks1 = imin(ks0 + CH, KSTEPS);
    if (ks0 >= ks1) return;

    const float* W; float* out; int N, n0;
    if (t < 32) { W = w512;  out = y1; N = 512; n0 = t * 16; }
    else        { W = w256b; out = z1; N = 256; n0 = (t - 32) * 16; }

    int h = lane >> 4, ln = lane & 15;
    const __bf16* xr = xact + (size_t)ln * FLATN;           // A row = batch ln
    const float* wc = W + n0 + ln;                          // B column base

    v8f acc = zero8f();
    for (int ks = ks0; ks < ks1; ++ks) {
        int k = ks * 32;
        v16bf a = load_pair(xr + k + h * 16, xr + k + h * 16 + 8);
        const float* wk = wc + (size_t)(k + h * 16) * N;
        V16 bt;
#pragma unroll
        for (int q = 0; q < 8; ++q)
            bt.p[q] = cvt2(wk[(size_t)(2 * q) * N], wk[(size_t)(2 * q + 1) * N]);
        if ((ks & 15) == 0 && (k + 544) < FLATN)
            __builtin_prefetch(&wc[(size_t)(k + 544) * N], 0, 1);
        acc = wmma_bf16(a, bt.v, acc);
    }
#pragma unroll
    for (int r = 0; r < 8; ++r)
        unsafeAtomicAdd(&out[(size_t)(r + h * 8) * N + n0 + ln], acc[r]);
}

// ---------------------------------------------------------------------------
// Generic small FC: out[16,N] = leaky( (leakyIn? leaky(in) : in) @ W + b ).
// One wave per 16-column tile, full K reduction in-wave.
// ---------------------------------------------------------------------------
__global__ __launch_bounds__(64) void fc_small_kernel(
    const float* __restrict__ in, const float* __restrict__ W,
    const float* __restrict__ bias, float* __restrict__ out,
    int K, int N, int leakyIn)
{
    int wave = (blockIdx.x * blockDim.x + threadIdx.x) >> 5;
    int lane = threadIdx.x & 31;
    int ntiles = (N + 15) >> 4;
    if (wave >= ntiles) return;
    int h = lane >> 4, ln = lane & 15;
    int col = wave * 16 + ln;
    int colc = imin(col, N - 1);
    const float* inr = in + (size_t)ln * K;

    v8f acc = zero8f();
    for (int k = 0; k < K; k += 32) {
        V16 a;
#pragma unroll
        for (int q = 0; q < 4; ++q) {
            float f0 = inr[k + h * 16 + 2 * q];
            float f1 = inr[k + h * 16 + 2 * q + 1];
            float f2 = inr[k + h * 16 + 8 + 2 * q];
            float f3 = inr[k + h * 16 + 8 + 2 * q + 1];
            if (leakyIn) {
                f0 = leakyf(f0); f1 = leakyf(f1);
                f2 = leakyf(f2); f3 = leakyf(f3);
            }
            a.p[q] = cvt2(f0, f1);
            a.p[4 + q] = cvt2(f2, f3);
        }
        V16 bt;
#pragma unroll
        for (int q = 0; q < 8; ++q)
            bt.p[q] = cvt2(W[(size_t)(k + h * 16 + 2 * q) * N + colc],
                           W[(size_t)(k + h * 16 + 2 * q + 1) * N + colc]);
        acc = wmma_bf16(a.v, bt.v, acc);
    }
    if (col < N) {
        float bb = bias[colc];
#pragma unroll
        for (int r = 0; r < 8; ++r)
            out[(size_t)(r + h * 8) * N + col] = leakyf(acc[r] + bb);
    }
}

// ---------------------------------------------------------------------------
// Heads: 225 softmax(13x13) heads and 225 sigmoid(4x4) heads (shared weights).
// ---------------------------------------------------------------------------
__global__ void heads_kernel(const float* __restrict__ y3,
                             const float* __restrict__ z5,
                             const float* __restrict__ wsm,
                             const float* __restrict__ bsm,
                             const float* __restrict__ wsig,
                             const float* __restrict__ bsig,
                             float* __restrict__ out)
{
    int idx = blockIdx.x * blockDim.x + threadIdx.x;
    if (idx < 3600) {
        int b = idx / 225, p = idx - b * 225;
        const float* yin = y3 + (size_t)b * 2925 + p * 13;
        float o[13], mx = -1e30f;
#pragma unroll
        for (int j = 0; j < 13; ++j) {
            float s = bsm[j];
            for (int i = 0; i < 13; ++i) s += yin[i] * wsm[i * 13 + j];
            o[j] = s; mx = fmaxf(mx, s);
        }
        float sum = 0.f;
#pragma unroll
        for (int j = 0; j < 13; ++j) { o[j] = __expf(o[j] - mx); sum += o[j]; }
        float inv = 1.f / sum;
#pragma unroll
        for (int j = 0; j < 13; ++j)
            out[(size_t)(b * 225 + p) * 13 + j] = o[j] * inv;
    } else if (idx < 7200) {
        int q = idx - 3600;
        int b = q / 225, p = q - b * 225;
        const float* zin = z5 + (size_t)b * 900 + p * 4;
#pragma unroll
        for (int j = 0; j < 4; ++j) {
            float s = bsig[j];
            for (int i = 0; i < 4; ++i) s += zin[i] * wsig[i * 4 + j];
            out[46800 + (size_t)(b * 225 + p) * 4 + j] = 1.f / (1.f + __expf(-s));
        }
    }
}

// ---------------------------------------------------------------------------
extern "C" void kernel_launch(void* const* d_in, const int* in_sizes, int n_in,
                              void* d_out, int out_size, void* d_ws, size_t ws_size,
                              hipStream_t stream)
{
    const float* hidden = (const float*)d_in[0];
    const float* w1     = (const float*)d_in[1];
    const float* b1     = (const float*)d_in[2];
    const float* w2     = (const float*)d_in[3];
    const float* b2     = (const float*)d_in[4];
    const float* w512   = (const float*)d_in[5];
    const float* b512   = (const float*)d_in[6];
    const float* w256a  = (const float*)d_in[7];
    const float* b256a  = (const float*)d_in[8];
    const float* wcls   = (const float*)d_in[9];
    const float* bcls   = (const float*)d_in[10];
    const float* wsm    = (const float*)d_in[11];
    const float* bsm    = (const float*)d_in[12];
    const float* w256b  = (const float*)d_in[13];
    const float* b256b  = (const float*)d_in[14];
    const float* w128a  = (const float*)d_in[15];
    const float* b128a  = (const float*)d_in[16];
    const float* w128b  = (const float*)d_in[17];
    const float* b128b  = (const float*)d_in[18];
    const float* w64    = (const float*)d_in[19];
    const float* b64    = (const float*)d_in[20];
    const float* wreg   = (const float*)d_in[21];
    const float* breg   = (const float*)d_in[22];
    const float* wsig   = (const float*)d_in[23];
    const float* bsig   = (const float*)d_in[24];
    float* outp = (float*)d_out;
    (void)in_sizes; (void)n_in; (void)out_size; (void)ws_size;

    char* ws = (char*)d_ws;
    size_t off = 0;
    auto wsalloc = [&](size_t bytes) -> char* {
        char* p = ws + off;
        off += (bytes + 255) & ~(size_t)255;
        return p;
    };
    __bf16* stage1 = (__bf16*)wsalloc((size_t)16 * 97 * 383 * 32 * 2);
    __bf16* xact   = (__bf16*)wsalloc((size_t)16 * FLATN * 2);
    float* y1 = (float*)wsalloc(16 * 512 * 4);
    float* z1 = (float*)wsalloc(16 * 256 * 4);
    float* y2 = (float*)wsalloc(16 * 256 * 4);
    float* y3 = (float*)wsalloc((size_t)16 * 2925 * 4);
    float* z2 = (float*)wsalloc(16 * 128 * 4);
    float* z3 = (float*)wsalloc(16 * 128 * 4);
    float* z4 = (float*)wsalloc(16 * 64 * 4);
    float* z5 = (float*)wsalloc((size_t)16 * 900 * 4);

    { int tasks = 16 * 97 * 48;
      conv1_pool_kernel<<<(tasks + 3) / 4, 128, 0, stream>>>(hidden, w1, b1, stage1); }
    { int tasks = 16 * 47 * 24;
      conv2_pool_kernel<<<(tasks + 3) / 4, 128, 0, stream>>>(stage1, w2, b2, xact); }
    { int n = 16 * 512 + 16 * 256;
      init_fc1_kernel<<<(n + 255) / 256, 256, 0, stream>>>(b512, b256b, y1, z1); }
    { int waves = 48 * 140;
      fc_big_kernel<<<(waves * 32 + 255) / 256, 256, 0, stream>>>(xact, w512, w256b, y1, z1); }

    auto small = [&](const float* in, const float* W, const float* bias,
                     float* out, int K, int N, int lin) {
        int ntiles = (N + 15) / 16;
        fc_small_kernel<<<(ntiles + 1) / 2, 64, 0, stream>>>(in, W, bias, out, K, N, lin);
    };
    small(y1, w256a, b256a, y2, 512, 256, 1);
    small(y2, wcls,  bcls,  y3, 256, 2925, 0);
    small(z1, w128a, b128a, z2, 256, 128, 1);
    small(z2, w128b, b128b, z3, 128, 128, 0);
    small(z3, w64,   b64,   z4, 128, 64, 0);
    small(z4, wreg,  breg,  z5, 64, 900, 0);

    heads_kernel<<<(7200 + 255) / 256, 256, 0, stream>>>(y3, z5, wsm, bsm, wsig, bsig, outp);
}
